// TelechatAttention_74706661146780
// MI455X (gfx1250) — matrix operand order repaired and checked by reading, not compiled, not run on hardware
//
#include <hip/hip_runtime.h>
#include <hip/hip_bf16.h>
#include <math.h>

// ---------------------------------------------------------------------------
// Telechat attention block for MI455X (gfx1250, wave32, WMMA).
// bf16 WMMA everywhere; one-time f32->bf16 conversion passes keep the GEMM
// inner loops free of conversion VALU.
// ---------------------------------------------------------------------------

#define H_DIM   5120
#define NHEADS  32
#define HEADD   160
#define SEQ     2048

typedef __attribute__((ext_vector_type(16))) __bf16 bf16x16;
typedef __attribute__((ext_vector_type(8)))  float  f32x8;

union BFrag { unsigned int u[8]; bf16x16 v; };
union Pack8 { unsigned short s[8]; uint4 q; };

__device__ __forceinline__ unsigned short f2bf(float f) {
    unsigned int u = __float_as_uint(f);
    unsigned int r = (u + 0x7FFFu + ((u >> 16) & 1u)) >> 16;  // RNE
    return (unsigned short)r;
}
__device__ __forceinline__ float bf2f(unsigned short h) {
    return __uint_as_float(((unsigned int)h) << 16);
}

// ---------------------------------------------------------------------------
// One-time f32 -> bf16 conversion (8 elements/thread, b128 in / b128 out).
// ---------------------------------------------------------------------------
__global__ __launch_bounds__(256)
void cvt_f32_bf16(const float* __restrict__ src, unsigned short* __restrict__ dst,
                  size_t n)
{
    const size_t i = ((size_t)blockIdx.x * 256 + threadIdx.x) * 8;
    if (i + 8 > n) return;
    const float4 f0 = *(const float4*)(src + i);
    const float4 f1 = *(const float4*)(src + i + 4);
    Pack8 p;
    p.s[0] = f2bf(f0.x); p.s[1] = f2bf(f0.y); p.s[2] = f2bf(f0.z); p.s[3] = f2bf(f0.w);
    p.s[4] = f2bf(f1.x); p.s[5] = f2bf(f1.y); p.s[6] = f2bf(f1.z); p.s[7] = f2bf(f1.w);
    *(uint4*)(dst + i) = p.q;
}

// ---------------------------------------------------------------------------
// C[M,N] = A[M,K] (bf16) * W[N,K]^T (bf16), f32 accumulate.
// epi==0 : write bf16 C
// epi==1 : write f32 C + bias[n] + resid[m*N+n]
// Tile 128x128, K-step 32, 256 threads = 8 waves in 2(M) x 4(N) grid;
// each wave: 64x32 sub-tile = 4x2 WMMA frags (8 WMMA / 6 frag loads / step).
// LDS double-buffered; global loads register-staged; 1 barrier per k-step.
// ---------------------------------------------------------------------------
__global__ __launch_bounds__(256)
void gemm_bf16_wmma(const unsigned short* __restrict__ A,
                    const unsigned short* __restrict__ W,
                    unsigned short* __restrict__ outBf, float* __restrict__ outF,
                    const float* __restrict__ bias, const float* __restrict__ resid,
                    int M, int N, int K, int epi)
{
    __shared__ unsigned short As[2][128][40];   // 128 x 32 bf16 (+pad, 80B rows)
    __shared__ unsigned short Ws[2][128][40];

    const int tid   = threadIdx.x;
    const int lane  = tid & 31;
    const int wid   = tid >> 5;
    const int waveM = wid >> 2;              // 0..1 -> 64 rows
    const int waveN = wid & 3;               // 0..3 -> 32 cols
    const int m0    = blockIdx.y * 128;
    const int n0    = blockIdx.x * 128;
    const int r16   = lane & 15;
    const int hi    = lane >> 4;

    f32x8 acc[4][2];
    const f32x8 zero = {0.f,0.f,0.f,0.f,0.f,0.f,0.f,0.f};
    #pragma unroll
    for (int i = 0; i < 4; i++) { acc[i][0] = zero; acc[i][1] = zero; }

    // Staging map: each thread copies 16 bf16 of one row (2 x b128).
    const int srow = tid >> 1;               // 0..127
    const int scol = (tid & 1) * 16;         // 0 / 16
    const unsigned short* aRow = A + (size_t)(m0 + srow) * K + scol;
    const unsigned short* wRow = W + (size_t)(n0 + srow) * K + scol;

    // Prologue: stage k0 = 0 into buffer 0.
    {
        const uint4 a0 = *(const uint4*)(aRow);
        const uint4 a1 = *(const uint4*)(aRow + 8);
        const uint4 w0 = *(const uint4*)(wRow);
        const uint4 w1 = *(const uint4*)(wRow + 8);
        *(uint4*)&As[0][srow][scol]     = a0;
        *(uint4*)&As[0][srow][scol + 8] = a1;
        *(uint4*)&Ws[0][srow][scol]     = w0;
        *(uint4*)&Ws[0][srow][scol + 8] = w1;
    }

    int buf = 0;
    for (int k0 = 0; k0 < K; k0 += 32) {
        __syncthreads();

        // Issue next tile's global loads early (register staged).
        const bool more = (k0 + 32) < K;
        uint4 a0, a1, w0, w1;
        if (more) {
            a0 = *(const uint4*)(aRow + k0 + 32);
            a1 = *(const uint4*)(aRow + k0 + 40);
            w0 = *(const uint4*)(wRow + k0 + 32);
            w1 = *(const uint4*)(wRow + k0 + 40);
            if (k0 + 64 < K) {                         // L2 prefetch 2 steps out
                __builtin_prefetch(aRow + k0 + 64, 0, 1);
                __builtin_prefetch(wRow + k0 + 64, 0, 1);
            }
        }

        // Fragment loads from current buffer.
        BFrag afr[4], bfr[2];
        #pragma unroll
        for (int fm = 0; fm < 4; fm++) {
            const int row = waveM * 64 + fm * 16 + r16;
            #pragma unroll
            for (int vv = 0; vv < 8; vv++) {
                const int kb = ((vv < 4) ? (2 * vv) : (16 + 2 * (vv - 4))) + hi * 8;
                afr[fm].u[vv] = *(const unsigned int*)&As[buf][row][kb];
            }
        }
        #pragma unroll
        for (int fn = 0; fn < 2; fn++) {
            const int col = waveN * 32 + fn * 16 + r16;
            #pragma unroll
            for (int vv = 0; vv < 8; vv++) {
                const int kb = ((vv < 4) ? (2 * vv) : (16 + 2 * (vv - 4))) + hi * 8;
                bfr[fn].u[vv] = *(const unsigned int*)&Ws[buf][col][kb];
            }
        }
        #pragma unroll
        for (int fm = 0; fm < 4; fm++)
            #pragma unroll
            for (int fn = 0; fn < 2; fn++)
                acc[fm][fn] = __builtin_amdgcn_wmma_f32_16x16x32_bf16(
                    false, afr[fm].v, false, bfr[fn].v,
                    (short)0, acc[fm][fn], false, false);

        // Commit next tile to the other buffer (waits only on its own loads).
        if (more) {
            *(uint4*)&As[buf ^ 1][srow][scol]     = a0;
            *(uint4*)&As[buf ^ 1][srow][scol + 8] = a1;
            *(uint4*)&Ws[buf ^ 1][srow][scol]     = w0;
            *(uint4*)&Ws[buf ^ 1][srow][scol + 8] = w1;
        }
        buf ^= 1;
    }

    // Epilogue. C layout: dword e -> M = e (+8 for hi lanes), N = lane&15.
    #pragma unroll
    for (int fm = 0; fm < 4; fm++) {
        #pragma unroll
        for (int fn = 0; fn < 2; fn++) {
            const int ncol = n0 + waveN * 32 + fn * 16 + r16;
            #pragma unroll
            for (int e = 0; e < 8; e++) {
                const int mrow = m0 + waveM * 64 + fm * 16 + e + hi * 8;
                const size_t idx = (size_t)mrow * N + ncol;
                const float val = acc[fm][fn][e];
                if (epi) outF[idx] = val + bias[ncol] + resid[idx];
                else     outBf[idx] = f2bf(val);
            }
        }
    }
}

// ---------------------------------------------------------------------------
// RoPE (NTK-scaled): S=2048 <= TRAIN_SEQLEN=8192 -> mscale=1, ntk_alpha=3,
// base = 10000 * 3^(160/158). Splits kv into K/V and re-lays Q/K/V as
// [head][s][d] bf16 for the attention kernel.
// ---------------------------------------------------------------------------
__global__ __launch_bounds__(HEADD)
void rope_kernel(const unsigned short* __restrict__ qtmp,
                 const unsigned short* __restrict__ kvtmp,
                 unsigned short* __restrict__ Q,
                 unsigned short* __restrict__ K,
                 unsigned short* __restrict__ V)
{
    const int s = blockIdx.x;
    const int n = blockIdx.y;
    const int d = threadIdx.x;                 // 0..159
    const int i = (d < 80) ? d : d - 80;

    const double base = 10000.0 * exp((160.0 / 158.0) * log(3.0));
    const double invf = exp(-((double)(2 * i) / 160.0) * log(base));
    const double ang  = (double)s * invf;
    const float c  = (float)cos(ang);
    const float sn = (float)sin(ang);

    const size_t qi = (size_t)s * H_DIM + (size_t)n * HEADD;
    const float qd = bf2f(qtmp[qi + d]);
    const float qo = (d < 80) ? -bf2f(qtmp[qi + d + 80]) : bf2f(qtmp[qi + d - 80]);

    const size_t ki = (size_t)s * (2 * H_DIM) + (size_t)n * (2 * HEADD);
    const float kd = bf2f(kvtmp[ki + d]);
    const float ko = (d < 80) ? -bf2f(kvtmp[ki + d + 80]) : bf2f(kvtmp[ki + d - 80]);

    const size_t o = ((size_t)n * SEQ + s) * HEADD + d;
    Q[o] = f2bf(qd * c + qo * sn);
    K[o] = f2bf(kd * c + ko * sn);
    V[o] = kvtmp[ki + HEADD + d];
}

// ---------------------------------------------------------------------------
// Causal flash attention, one (head, 32-row q-tile) per workgroup.
// 128 threads = 4 waves. Q/K tiles 32x160 bf16, V transposed 160x32 bf16,
// scores 32x32 f32, online softmax, O accumulated 32x160 f32 in LDS.
// QK^T: 1 frag/wave x 5 K-steps. PV: 5 frags/wave x 1 K-step (K=32 keys).
// ctx written bf16 for the output projection GEMM. Static LDS ~60KB.
// ---------------------------------------------------------------------------
__global__ __launch_bounds__(128)
void flash_attn_kernel(const unsigned short* __restrict__ Q,
                       const unsigned short* __restrict__ K,
                       const unsigned short* __restrict__ V,
                       unsigned short* __restrict__ ctx)
{
    __shared__ unsigned short Qs[32][160];
    __shared__ unsigned short Ks[32][160];
    __shared__ unsigned short Vt[160][36];     // [head-dim][key] (+pad)
    __shared__ float          Sc[32][33];
    __shared__ unsigned short Ps[32][36];
    __shared__ float          Os[32][164];
    __shared__ float mArr[32], lArr[32], aArr[32];

    const int qb  = blockIdx.x;                // q tile (32 rows)
    const int hn  = blockIdx.y;                // head
    const int tid = threadIdx.x;
    const int lane = tid & 31;
    const int wid  = tid >> 5;                 // 0..3
    const int r16  = lane & 15;
    const int hi   = lane >> 4;
    const float inv_norm = 0.07905694150420949f;   // 1/sqrt(160)

    const unsigned short* Qh = Q + ((size_t)hn * SEQ) * HEADD;
    const unsigned short* Kh = K + ((size_t)hn * SEQ) * HEADD;
    const unsigned short* Vh = V + ((size_t)hn * SEQ) * HEADD;

    // Load Q tile (contiguous 5120 bf16) and init state.
    {
        const unsigned int* src = (const unsigned int*)(Qh + (size_t)qb * 32 * HEADD);
        unsigned int* dst = (unsigned int*)&Qs[0][0];
        for (int t = tid; t < 32 * HEADD / 2; t += 128) dst[t] = src[t];
    }
    for (int t = tid; t < 32 * HEADD; t += 128) Os[t / HEADD][t % HEADD] = 0.f;
    if (tid < 32) { mArr[tid] = -3.0e38f; lArr[tid] = 0.f; }
    __syncthreads();

    for (int kb = 0; kb <= qb; kb++) {
        // Stage K tile + transposed V tile.
        {
            const unsigned int* src = (const unsigned int*)(Kh + (size_t)kb * 32 * HEADD);
            unsigned int* dst = (unsigned int*)&Ks[0][0];
            for (int t = tid; t < 32 * HEADD / 2; t += 128) dst[t] = src[t];
        }
        {
            const unsigned short* vsrc = Vh + (size_t)kb * 32 * HEADD;
            for (int t = tid; t < 32 * HEADD; t += 128) {
                const int key = t / HEADD, d = t % HEADD;
                Vt[d][key] = vsrc[t];
            }
        }
        __syncthreads();

        // ---- S = Q * K^T  (one 16x16 frag per wave, 5 K-steps over head-dim)
        const int waveM = wid & 1;             // q-row half
        const int waveN = wid >> 1;            // key-col half
        f32x8 sfr = {0.f,0.f,0.f,0.f,0.f,0.f,0.f,0.f};
        #pragma unroll
        for (int kk = 0; kk < 5; kk++) {
            BFrag a, b;
            const int arow = waveM * 16 + r16;
            #pragma unroll
            for (int vv = 0; vv < 8; vv++) {
                const int kbh = kk * 32 + ((vv < 4) ? (2 * vv) : (16 + 2 * (vv - 4))) + hi * 8;
                a.u[vv] = *(const unsigned int*)&Qs[arow][kbh];
            }
            const int bcol = waveN * 16 + r16;
            #pragma unroll
            for (int vv = 0; vv < 8; vv++) {
                const int kbh = kk * 32 + ((vv < 4) ? (2 * vv) : (16 + 2 * (vv - 4))) + hi * 8;
                b.u[vv] = *(const unsigned int*)&Ks[bcol][kbh];
            }
            sfr = __builtin_amdgcn_wmma_f32_16x16x32_bf16(
                false, a.v, false, b.v, (short)0, sfr, false, false);
        }
        // Scale + causal mask, spill to LDS scores.
        #pragma unroll
        for (int e = 0; e < 8; e++) {
            const int m = waveM * 16 + e + hi * 8;
            const int n = waveN * 16 + r16;
            float val = sfr[e] * inv_norm;
            if (kb * 32 + n > qb * 32 + m) val = -1.0e30f;
            Sc[m][n] = val;
        }
        __syncthreads();

        // ---- Online softmax (one thread per q row).
        if (tid < 32) {
            const int r = tid;
            float mOld = mArr[r];
            float mx = mOld;
            #pragma unroll 4
            for (int j = 0; j < 32; j++) mx = fmaxf(mx, Sc[r][j]);
            const float alpha = __expf(mOld - mx);
            float sum = 0.f;
            #pragma unroll 4
            for (int j = 0; j < 32; j++) {
                const float p = __expf(Sc[r][j] - mx);
                sum += p;
                Ps[r][j] = f2bf(p);
            }
            mArr[r] = mx;
            lArr[r] = alpha * lArr[r] + sum;
            aArr[r] = alpha;
        }
        __syncthreads();

        // Rescale running O by alpha.
        for (int t = tid; t < 32 * HEADD; t += 128) {
            const int r = t / HEADD, d = t % HEADD;
            Os[r][d] *= aArr[r];
        }
        __syncthreads();

        // ---- O += P * V  (per wave: 16 q-rows x 80 head-dims = 5 frags, K=32)
        const int pvM = wid & 1;               // q-row half
        const int pvN = wid >> 1;              // head-dim half (80 each)
        #pragma unroll
        for (int f = 0; f < 5; f++) {
            BFrag a, b;
            const int arow = pvM * 16 + r16;
            #pragma unroll
            for (int vv = 0; vv < 8; vv++) {
                const int kbh = ((vv < 4) ? (2 * vv) : (16 + 2 * (vv - 4))) + hi * 8;
                a.u[vv] = *(const unsigned int*)&Ps[arow][kbh];
            }
            const int ncol = pvN * 80 + f * 16 + r16;
            #pragma unroll
            for (int vv = 0; vv < 8; vv++) {
                const int kbh = ((vv < 4) ? (2 * vv) : (16 + 2 * (vv - 4))) + hi * 8;
                b.u[vv] = *(const unsigned int*)&Vt[ncol][kbh];
            }
            f32x8 pv = {0.f,0.f,0.f,0.f,0.f,0.f,0.f,0.f};
            pv = __builtin_amdgcn_wmma_f32_16x16x32_bf16(
                false, a.v, false, b.v, (short)0, pv, false, false);
            #pragma unroll
            for (int e = 0; e < 8; e++) {
                const int m = pvM * 16 + e + hi * 8;
                Os[m][ncol] += pv[e];
            }
        }
        __syncthreads();
    }

    // Normalize and write ctx [s][h] bf16.
    for (int t = tid; t < 32 * HEADD; t += 128) {
        const int r = t / HEADD, d = t % HEADD;
        const float o = Os[r][d] / lArr[r];
        ctx[(size_t)(qb * 32 + r) * H_DIM + (size_t)hn * HEADD + d] = f2bf(o);
    }
}

// ---------------------------------------------------------------------------
// Host-side orchestration.
// ---------------------------------------------------------------------------
extern "C" void kernel_launch(void* const* d_in, const int* in_sizes, int n_in,
                              void* d_out, int out_size, void* d_ws, size_t ws_size,
                              hipStream_t stream) {
    (void)in_sizes; (void)n_in; (void)out_size; (void)ws_size;

    const float* hidden = (const float*)d_in[0];
    const float* resid  = (const float*)d_in[1];
    /* d_in[2] = attention_mask: causal, recomputed analytically */
    const float* Wq     = (const float*)d_in[3];
    const float* Wkv    = (const float*)d_in[4];
    const float* Wd     = (const float*)d_in[5];
    const float* bd     = (const float*)d_in[6];
    float* out          = (float*)d_out;

    // Workspace layout (bytes):
    char* ws = (char*)d_ws;
    unsigned short* hbf   = (unsigned short*)(ws);                    // 20.97 MB
    unsigned short* Wqbf  = (unsigned short*)(ws +  20971520ull);     // 52.43 MB
    unsigned short* Wkvbf = (unsigned short*)(ws +  73400320ull);     // 104.86 MB
    unsigned short* Wdbf  = (unsigned short*)(ws + 178257920ull);     // 52.43 MB
    unsigned short* qtmp  = (unsigned short*)(ws + 230686720ull);     // 20.97 MB
    unsigned short* kvtmp = (unsigned short*)(ws + 251658240ull);     // 41.94 MB
    unsigned short* Qw    = (unsigned short*)(ws + 293601280ull);     // 20.97 MB
    unsigned short* Kw    = (unsigned short*)(ws + 314572800ull);
    unsigned short* Vw    = (unsigned short*)(ws + 335544320ull);
    unsigned short* ctxbf = (unsigned short*)(ws + 356515840ull);     // 20.97 MB

    // 0) One-time f32 -> bf16 conversions.
    cvt_f32_bf16<<<  5120, 256, 0, stream>>>(hidden, hbf,   (size_t)SEQ * H_DIM);
    cvt_f32_bf16<<< 12800, 256, 0, stream>>>(Wq,     Wqbf,  (size_t)H_DIM * H_DIM);
    cvt_f32_bf16<<< 25600, 256, 0, stream>>>(Wkv,    Wkvbf, (size_t)2 * H_DIM * H_DIM);
    cvt_f32_bf16<<< 12800, 256, 0, stream>>>(Wd,     Wdbf,  (size_t)H_DIM * H_DIM);

    // 1) q = hidden @ Wq^T            (2048 x 5120 x 5120)
    gemm_bf16_wmma<<<dim3(H_DIM / 128, SEQ / 128), 256, 0, stream>>>(
        hbf, Wqbf, qtmp, nullptr, nullptr, nullptr, SEQ, H_DIM, H_DIM, 0);

    // 2) kv = hidden @ Wkv^T          (2048 x 10240 x 5120)
    gemm_bf16_wmma<<<dim3(2 * H_DIM / 128, SEQ / 128), 256, 0, stream>>>(
        hbf, Wkvbf, kvtmp, nullptr, nullptr, nullptr, SEQ, 2 * H_DIM, H_DIM, 0);

    // 3) RoPE + head-major relayout
    rope_kernel<<<dim3(SEQ, NHEADS), HEADD, 0, stream>>>(qtmp, kvtmp, Qw, Kw, Vw);

    // 4) Causal flash attention
    flash_attn_kernel<<<dim3(SEQ / 32, NHEADS), 128, 0, stream>>>(Qw, Kw, Vw, ctxbf);

    // 5) out = residual + ctx @ Wd^T + bd
    gemm_bf16_wmma<<<dim3(H_DIM / 128, SEQ / 128), 256, 0, stream>>>(
        ctxbf, Wdbf, nullptr, out, bd, resid, SEQ, H_DIM, H_DIM, 1);
}
